// GNN_36979668419077
// MI455X (gfx1250) — matrix operand order repaired and checked by reading, not compiled, not run on hardware
//
#include <hip/hip_runtime.h>
#include <hip/hip_bf16.h>
#include <math.h>
#include <stdint.h>

typedef __attribute__((ext_vector_type(16))) __bf16 v16bf;
typedef __attribute__((ext_vector_type(8)))  float  v8f;

#define BM 64
#define BN 128
#define BK 32

// ---------------------------------------------------------------------------
// Edge scatter-add:  Agg[dst] += H[src]   (4 features per thread, float4 load)
// ---------------------------------------------------------------------------
__global__ void gin_agg_scatter(const float* __restrict__ Hin,
                                const int* __restrict__ src,
                                const int* __restrict__ dst,
                                float* __restrict__ Agg,
                                int E, int Feat) {
    int e = blockIdx.x * blockDim.x + threadIdx.x;
    if (e >= E) return;
    int f = blockIdx.y * 4;
    int s = src[e], d = dst[e];
    const float4 val = *(const float4*)(Hin + (size_t)s * Feat + f);
    float* op = Agg + (size_t)d * Feat + f;
    atomicAdd(op + 0, val.x);
    atomicAdd(op + 1, val.y);
    atomicAdd(op + 2, val.z);
    atomicAdd(op + 3, val.w);
}

// ---------------------------------------------------------------------------
// fp32 -> bf16 with zero padding into [Rtot, KP] (Rtot = gridDim.x)
// ---------------------------------------------------------------------------
__global__ void gin_cvt_pad_bf16(const float* __restrict__ in,
                                 __bf16* __restrict__ out,
                                 int Rin, int Kin, int KP) {
    int r = blockIdx.x;
    int c = blockIdx.y * blockDim.x + threadIdx.x;
    if (c >= KP) return;
    float v = (r < Rin && c < Kin) ? in[(size_t)r * Kin + c] : 0.f;
    out[(size_t)r * KP + c] = (__bf16)v;
}

// ---------------------------------------------------------------------------
// WMMA GEMM on padded operands:
//   C[N,H] = A[NP,KP] * W[HP,KP]^T + bias
// Double-buffered GLOBAL_LOAD_ASYNC_TO_LDS_B128 staging (ASYNCcnt) so the DMA
// for k-step s+1 overlaps the ds_load_b128 -> v_wmma compute of k-step s.
// Block: 256 threads = 8 waves; block tile 64x128; k-step 32.
// ---------------------------------------------------------------------------
__global__ __launch_bounds__(256)
void gin_gemm_bf16_wmma(const __bf16* __restrict__ A,
                        const __bf16* __restrict__ W,
                        const float* __restrict__ bias,
                        float* __restrict__ Cout,
                        int Nrows, int KP, int Hc) {
    __shared__ uint32_t As[2][BM][16];   // 2 x 4 KB
    __shared__ uint32_t Ws[2][BN][16];   // 2 x 8 KB

    const int tid  = threadIdx.x;
    const int lane = tid & 31;
    const int wave = tid >> 5;     // 0..7
    const int wrow = wave >> 2;    // 0..1 -> 32-row strip
    const int wcol = wave & 3;     // 0..3 -> 32-col strip
    const int block_row = blockIdx.y * BM;
    const int block_col = blockIdx.x * BN;
    const int half = lane >> 4;
    const int mrow = lane & 15;

    // async staging lane mapping: one b128 (16 B = 8 bf16) per thread per tile
    const int r    = tid >> 2;     // 0..63
    const int part = tid & 3;      // 0..3 -> 16B chunk within a 64B row slice

    const uint32_t lane_off = (uint32_t)(r * 64 + part * 16);
    uint32_t aoff[2], woff[2], woff2[2];
    aoff[0]  = (uint32_t)(uintptr_t)(void*)&As[0][0][0] + lane_off;
    aoff[1]  = (uint32_t)(uintptr_t)(void*)&As[1][0][0] + lane_off;
    woff[0]  = (uint32_t)(uintptr_t)(void*)&Ws[0][0][0] + lane_off;
    woff[1]  = (uint32_t)(uintptr_t)(void*)&Ws[1][0][0] + lane_off;
    woff2[0] = woff[0] + 64u * 64u;
    woff2[1] = woff[1] + 64u * 64u;

    const __bf16* ga  = A + (size_t)(block_row + r) * KP + part * 8;
    const __bf16* gw0 = W + (size_t)(block_col + r) * KP + part * 8;
    const __bf16* gw1 = W + (size_t)(block_col + 64 + r) * KP + part * 8;

    auto issue = [&](int buf, int k0) {
        asm volatile("global_load_async_to_lds_b128 %0, %1, off"
                     :: "v"(aoff[buf]),  "v"(ga + k0)  : "memory");
        asm volatile("global_load_async_to_lds_b128 %0, %1, off"
                     :: "v"(woff[buf]),  "v"(gw0 + k0) : "memory");
        asm volatile("global_load_async_to_lds_b128 %0, %1, off"
                     :: "v"(woff2[buf]), "v"(gw1 + k0) : "memory");
    };

    v8f acc[2][2] = {};

    // prologue: fill buffer 0
    issue(0, 0);
    asm volatile("s_wait_asynccnt 0" ::: "memory");
    __syncthreads();

    const int nk = KP / BK;
    for (int ks = 0; ks < nk; ++ks) {
        const int cur = ks & 1;
        if (ks + 1 < nk) issue(cur ^ 1, (ks + 1) * BK);   // overlap DMA w/ compute

#pragma unroll
        for (int i = 0; i < 2; ++i) {
            // A fragment (16x32 bf16): lane half selects K phase (+8 / +24)
            union { uint32_t u[8]; v16bf v; } af;
            int ar = wrow * 32 + i * 16 + mrow;
#pragma unroll
            for (int q = 0; q < 8; ++q) {
                int p = (q < 4) ? (4 * half + q) : (8 + 4 * half + (q - 4));
                af.u[q] = As[cur][ar][p];
            }
#pragma unroll
            for (int j = 0; j < 2; ++j) {
                // B fragment (32x16 bf16): lane = N col, VGPR q = K pair
                union { uint32_t u[8]; v16bf v; } bfr;
                int bc = wcol * 32 + j * 16 + mrow;
#pragma unroll
                for (int q = 0; q < 8; ++q)
                    bfr.u[q] = Ws[cur][bc][(q < 4) ? (4 * half + q) : (8 + 4 * half + (q - 4))];

                acc[i][j] = __builtin_amdgcn_wmma_f32_16x16x32_bf16(
                    false, af.v, false, bfr.v, (short)0, acc[i][j], false, false);
            }
        }

        asm volatile("s_wait_asynccnt 0" ::: "memory");
        __syncthreads();
    }

    // epilogue: + bias, masked fp32 store to the real [N,H] output
#pragma unroll
    for (int i = 0; i < 2; ++i)
#pragma unroll
        for (int j = 0; j < 2; ++j) {
            int col = block_col + wcol * 32 + j * 16 + mrow;
            if (col >= Hc) continue;
            float b = bias[col];
            int row0 = block_row + wrow * 32 + i * 16 + 8 * half;
            float* cp = Cout + (size_t)row0 * Hc + col;
#pragma unroll
            for (int q = 0; q < 8; ++q) {
                if (row0 + q < Nrows) *cp = acc[i][j][q] + b;
                cp += Hc;
            }
        }
}

// ---------------------------------------------------------------------------
// Column statistics: per-column sum and sum-of-squares (atomics into [H])
// ---------------------------------------------------------------------------
__global__ __launch_bounds__(256)
void gin_colstats(const float* __restrict__ Z, float* __restrict__ csum,
                  float* __restrict__ csq, int Nrows, int Hc) {
    int row0 = blockIdx.x * 64;
    int rend = min(row0 + 64, Nrows);
    for (int c = threadIdx.x; c < Hc; c += 256) {
        float s = 0.f, q = 0.f;
        for (int r = row0; r < rend; ++r) {
            float z = Z[(size_t)r * Hc + c];
            s += z;
            q += z * z;
        }
        atomicAdd(&csum[c], s);
        atomicAdd(&csq[c], q);
    }
}

// ---------------------------------------------------------------------------
// Fused BN (training stats) + ReLU + cast to padded bf16 [NP, KP]
// ---------------------------------------------------------------------------
__global__ void gin_bn_relu_pad_bf16(const float* __restrict__ Z,
                                     const float* __restrict__ csum,
                                     const float* __restrict__ csq,
                                     const float* __restrict__ g,
                                     const float* __restrict__ be,
                                     __bf16* __restrict__ out,
                                     int Nrows, int Hc, int KP, float invN) {
    int r = blockIdx.x;
    int c = blockIdx.y * blockDim.x + threadIdx.x;
    if (c >= KP) return;
    float val = 0.f;
    if (r < Nrows && c < Hc) {
        float m  = csum[c] * invN;
        float v  = csq[c] * invN - m * m;
        float sc = g[c] * rsqrtf(v + 1e-5f);
        float sh = be[c] - m * sc;
        val = fmaxf(Z[(size_t)r * Hc + c] * sc + sh, 0.f);
    }
    out[(size_t)r * KP + c] = (__bf16)val;
}

// ---------------------------------------------------------------------------
// Graph pooling: pools[batch[n], off + c] += h[n, c]
// ---------------------------------------------------------------------------
__global__ void gin_pool_scatter(const float* __restrict__ Hf,
                                 const int* __restrict__ batch,
                                 float* __restrict__ pools,
                                 int Hc, int poolStride, int off) {
    int c = blockIdx.y * blockDim.x + threadIdx.x;
    if (c >= Hc) return;
    int r = blockIdx.x;
    atomicAdd(&pools[(size_t)batch[r] * poolStride + off + c],
              Hf[(size_t)r * Hc + c]);
}

// ---------------------------------------------------------------------------
// Readout head linear (small): out[g,o] = relu?( X[g,:] . W[o,:] + b[o] )
// ---------------------------------------------------------------------------
__global__ void gin_head_lin(const float* __restrict__ X,
                             const float* __restrict__ Wm,
                             const float* __restrict__ b,
                             float* __restrict__ out,
                             int Kin, int Hout, int do_relu) {
    int o = blockIdx.y * blockDim.x + threadIdx.x;
    if (o >= Hout) return;
    int gi = blockIdx.x;
    const float* xr = X + (size_t)gi * Kin;
    const float* wr = Wm + (size_t)o * Kin;
    float s = b[o];
    for (int k = 0; k < Kin; ++k) s = fmaf(xr[k], wr[k], s);
    out[(size_t)gi * Hout + o] = do_relu ? fmaxf(s, 0.f) : s;
}

__global__ void gin_log_softmax(const float* __restrict__ logits,
                                float* __restrict__ out, int Gn, int Cn) {
    int gi = blockIdx.x * blockDim.x + threadIdx.x;
    if (gi >= Gn) return;
    const float* lr = logits + (size_t)gi * Cn;
    float mx = -INFINITY;
    for (int c = 0; c < Cn; ++c) mx = fmaxf(mx, lr[c]);
    float se = 0.f;
    for (int c = 0; c < Cn; ++c) se += expf(lr[c] - mx);
    float lse = logf(se) + mx;
    for (int c = 0; c < Cn; ++c) out[(size_t)gi * Cn + c] = lr[c] - lse;
}

// ---------------------------------------------------------------------------
extern "C" void kernel_launch(void* const* d_in, const int* in_sizes, int n_in,
                              void* d_out, int out_size, void* d_ws, size_t ws_size,
                              hipStream_t stream) {
    const int N = in_sizes[2];                 // 50000
    const int E = in_sizes[1] / 2;             // 400000
    const int H = in_sizes[5];                 // 400
    const int M = in_sizes[0] / N;             // 4
    const int C = in_sizes[27];                // 10
    const int L = in_sizes[14] / (H * H);      // 2
    const int D = L + 1;                       // 3
    const int G = 128;                         // num_graphs (reference constant)
    const int H1 = in_sizes[25];               // 600
    const int PH = D * H;                      // 1200

    // padded geometry (tiles always full inside the GEMM)
    const int NP  = (N + BM - 1) & ~(BM - 1);  // 50048
    const int HP  = (H + BN - 1) & ~(BN - 1);  // 512
    const int KPH = (H + BK - 1) & ~(BK - 1);  // 416
    const int KPM = (M + BK - 1) & ~(BK - 1);  // 32

    const float* x      = (const float*)d_in[0];
    const int*   ei     = (const int*)d_in[1];
    const int*   src    = ei;
    const int*   dst    = ei + E;
    const int*   batch  = (const int*)d_in[2];
    const float* c0_w1  = (const float*)d_in[4];
    const float* c0_b1  = (const float*)d_in[5];
    const float* c0_g1  = (const float*)d_in[6];
    const float* c0_be1 = (const float*)d_in[7];
    const float* c0_w2  = (const float*)d_in[8];
    const float* c0_b2  = (const float*)d_in[9];
    const float* c0_g2  = (const float*)d_in[10];
    const float* c0_be2 = (const float*)d_in[11];
    const float* c0_w3  = (const float*)d_in[12];
    const float* c0_b3  = (const float*)d_in[13];
    const float* cw1    = (const float*)d_in[14];
    const float* cb1    = (const float*)d_in[15];
    const float* cg1    = (const float*)d_in[16];
    const float* cbe1   = (const float*)d_in[17];
    const float* cw2    = (const float*)d_in[18];
    const float* cb2    = (const float*)d_in[19];
    const float* cg2    = (const float*)d_in[20];
    const float* cbe2   = (const float*)d_in[21];
    const float* cw3    = (const float*)d_in[22];
    const float* cb3    = (const float*)d_in[23];
    const float* lin1_w = (const float*)d_in[24];
    const float* lin1_b = (const float*)d_in[25];
    const float* lin2_w = (const float*)d_in[26];
    const float* lin2_b = (const float*)d_in[27];

    // carve workspace
    char* wsp = (char*)d_ws;
    auto carve = [&](size_t bytes) {
        char* p = wsp;
        wsp += (bytes + 255) & ~(size_t)255;
        return p;
    };
    float*  AGG    = (float*)carve((size_t)N * H * 4);
    float*  Hbuf   = (float*)carve((size_t)N * H * 4);
    float*  Z      = (float*)carve((size_t)N * H * 4);
    __bf16* ABF    = (__bf16*)carve((size_t)NP * KPH * 2);
    __bf16* WBF    = (__bf16*)carve((size_t)HP * KPH * 2);
    float*  ssum   = (float*)carve((size_t)H * 4);
    float*  ssq    = (float*)carve((size_t)H * 4);
    float*  pools  = (float*)carve((size_t)G * PH * 4);
    float*  t1     = (float*)carve((size_t)G * H1 * 4);
    float*  logits = (float*)carve((size_t)G * C * 4);

    auto agg = [&](const float* hin, int Feat) {
        hipMemsetAsync(AGG, 0, (size_t)N * Feat * 4, stream);
        dim3 grid((E + 255) / 256, Feat / 4);
        gin_agg_scatter<<<grid, dim3(256), 0, stream>>>(hin, src, dst, AGG, E, Feat);
    };
    // pad-convert AGG -> ABF
    auto cvtA = [&](int K, int KP) {
        dim3 grid(NP, (KP + 255) / 256);
        gin_cvt_pad_bf16<<<grid, dim3(256), 0, stream>>>(AGG, ABF, N, K, KP);
    };
    auto gemm = [&](int K, const float* wsrc, const float* bias, float* Cout) {
        int KP = (K + BK - 1) & ~(BK - 1);
        dim3 gw(HP, (KP + 255) / 256);
        gin_cvt_pad_bf16<<<gw, dim3(256), 0, stream>>>(wsrc, WBF, H, K, KP);
        dim3 grid(HP / BN, NP / BM);
        gin_gemm_bf16_wmma<<<grid, dim3(256), 0, stream>>>(ABF, WBF, bias, Cout,
                                                           N, KP, H);
    };
    auto bn = [&](const float* g, const float* be) {
        hipMemsetAsync(ssum, 0, (size_t)H * 4, stream);
        hipMemsetAsync(ssq, 0, (size_t)H * 4, stream);
        gin_colstats<<<dim3((N + 63) / 64), dim3(256), 0, stream>>>(Z, ssum, ssq, N, H);
        dim3 grid(NP, (KPH + 255) / 256);
        gin_bn_relu_pad_bf16<<<grid, dim3(256), 0, stream>>>(
            Z, ssum, ssq, g, be, ABF, N, H, KPH, 1.0f / (float)N);
    };
    auto pool = [&](int off) {
        dim3 grid(N, (H + 255) / 256);
        gin_pool_scatter<<<grid, dim3(256), 0, stream>>>(Hbuf, batch, pools, H, PH, off);
    };

    hipMemsetAsync(pools, 0, (size_t)G * PH * 4, stream);

    // ---- layer 0 : M -> H -> H -> H -----------------------------------
    agg(x, M);
    cvtA(M, KPM);
    gemm(M, c0_w1, c0_b1, Z);
    bn(c0_g1, c0_be1);
    gemm(H, c0_w2, c0_b2, Z);
    bn(c0_g2, c0_be2);
    gemm(H, c0_w3, c0_b3, Hbuf);
    pool(0);

    // ---- layers 1 .. D-1 ----------------------------------------------
    for (int i = 0; i < L; ++i) {
        agg(Hbuf, H);
        cvtA(H, KPH);
        gemm(H, cw1 + (size_t)i * H * H, cb1 + (size_t)i * H, Z);
        bn(cg1 + (size_t)i * H, cbe1 + (size_t)i * H);
        gemm(H, cw2 + (size_t)i * H * H, cb2 + (size_t)i * H, Z);
        bn(cg2 + (size_t)i * H, cbe2 + (size_t)i * H);
        gemm(H, cw3 + (size_t)i * H * H, cb3 + (size_t)i * H, Hbuf);
        pool((i + 1) * H);
    }

    // ---- readout head --------------------------------------------------
    {
        dim3 g1(G, (H1 + 255) / 256);
        gin_head_lin<<<g1, dim3(256), 0, stream>>>(pools, lin1_w, lin1_b, t1, PH, H1, 1);
        dim3 g2(G, (C + 255) / 256);
        gin_head_lin<<<g2, dim3(256), 0, stream>>>(t1, lin2_w, lin2_b, logits, H1, C, 0);
        gin_log_softmax<<<dim3((G + 255) / 256), dim3(256), 0, stream>>>(
            logits, (float*)d_out, G, C);
    }
}
